// ConvBlock_50680614093178
// MI455X (gfx1250) — matrix operand order repaired and checked by reading, not compiled, not run on hardware
//
#include <hip/hip_runtime.h>
#include <hip/hip_bf16.h>
#include <stdint.h>

typedef __attribute__((ext_vector_type(16))) _Float16 v16h;
typedef __attribute__((ext_vector_type(8)))  float    v8f;

#define BATCH 4
#define CIN   64
#define CO    128
#define LEN   8192
#define LPAD  8200
#define L2    4096
#define DI    256
#define DM    128
#define DSTATE 16

// ---------------------------------------------------------------------------
// Weight-norm + f32->f16 conversion: one block per output channel.
// w = v * g[co] / ||v[co]||,   v is [CO][KD]
__global__ void wnorm_kernel(const float* __restrict__ v, const float* __restrict__ g,
                             _Float16* __restrict__ out, int KD) {
    __shared__ float red[64];
    __shared__ float scale_s;
    int co = blockIdx.x, tid = threadIdx.x;
    float s = 0.f;
    for (int k = tid; k < KD; k += 64) { float x = v[(size_t)co * KD + k]; s += x * x; }
    red[tid] = s; __syncthreads();
    for (int o = 32; o > 0; o >>= 1) { if (tid < o) red[tid] += red[tid + o]; __syncthreads(); }
    if (tid == 0) scale_s = g[co] / sqrtf(red[0]);
    __syncthreads();
    float sc = scale_s;
    for (int k = tid; k < KD; k += 64)
        out[(size_t)co * KD + k] = (_Float16)(v[(size_t)co * KD + k] * sc);
}

// generic f32 -> f16
__global__ void cvt_kernel(const float* __restrict__ s, _Float16* __restrict__ d, int n) {
    int i = blockIdx.x * blockDim.x + threadIdx.x;
    if (i < n) d[i] = (_Float16)s[i];
}

// x_proj (2,40,256) -> padded (2,48,256) f16, rows 40..47 zero
__global__ void cvt_xproj_kernel(const float* __restrict__ s, _Float16* __restrict__ d) {
    int i = blockIdx.x * blockDim.x + threadIdx.x;
    int n = 2 * 48 * 256;
    if (i >= n) return;
    int c = i % 256, q = (i / 256) % 48, m = i / (256 * 48);
    d[i] = (q < 40) ? (_Float16)s[((size_t)m * 40 + q) * 256 + c] : (_Float16)0.f;
}

// x (B,CIN,LEN) f32 -> (B,CIN,LPAD) f16, right zero pad
__global__ void cvt_xpad_kernel(const float* __restrict__ x, _Float16* __restrict__ xh) {
    int i = blockIdx.x * blockDim.x + threadIdx.x;
    int n = BATCH * CIN * LPAD;
    if (i >= n) return;
    int l = i % LPAD, c = (i / LPAD) % CIN, b = i / (LPAD * CIN);
    xh[i] = (l < LEN) ? (_Float16)x[((size_t)b * CIN + c) * LEN + l] : (_Float16)0.f;
}

// ---------------------------------------------------------------------------
// Fragment helpers (lane-private gathers in this code's contiguous k order, same for A and B).
// A (16-bit, 16x32): per-lane halfword j -> k = j + 16*h
// B (16-bit, 32x16): per-lane halfword j -> k = j + 16*h
__device__ __forceinline__ v16h load_a_frag(const _Float16* __restrict__ arow, int hl) {
    union { v16h v; uint32_t w[8]; } a;
#pragma unroll
    for (int jj = 0; jj < 8; ++jj)
        a.w[jj] = *(const uint32_t*)(arow + 2 * jj + 16 * hl);
    return a.v;
}
__device__ __forceinline__ v16h load_b_frag(const _Float16* __restrict__ brow, int hl) {
    union { v16h v; uint32_t w[8]; } bf;
#pragma unroll
    for (int jj = 0; jj < 8; ++jj)
        bf.w[jj] = *(const uint32_t*)(brow + 2 * jj + 16 * hl);
    return bf.v;
}

// ---------------------------------------------------------------------------
// Implicit-GEMM conv1d (valid, input pre-padded) via WMMA f16->f32.
// out[b][m][l] = sum_{ci,k} W[m][ci*8+k] * X[b][ci][l+k]
// grid (LEN/64, 1, B), block 256: 8 waves -> 8 m-tiles (Co=128);
// each wave computes 4 n-tiles so the A fragment is loaded once per 4 WMMAs.
__global__ __launch_bounds__(256)
void conv_wmma_kernel(const _Float16* __restrict__ W, const _Float16* __restrict__ X,
                      float* __restrict__ out, int Cin, int KD) {
    int b = blockIdx.z;
    int n0 = blockIdx.x * 64;
    int wave = threadIdx.x >> 5, lane = threadIdx.x & 31;
    int m0 = wave * 16;
    int nl = lane & 15, hl = lane >> 4;
    v8f acc[4] = {{}, {}, {}, {}};
    const _Float16* arow = W + (size_t)(m0 + nl) * KD;
    for (int kc = 0; kc < KD; kc += 32) {
        __builtin_prefetch(arow + kc + 64, 0, 1);
        v16h a = load_a_frag(arow + kc, hl);
#pragma unroll
        for (int u = 0; u < 4; ++u) {
            int l = n0 + 16 * u + nl;
            union { v16h v; _Float16 h[16]; } bf;
#pragma unroll
            for (int jj = 0; jj < 8; ++jj) {
                int kk = kc + 2 * jj + 16 * hl;   // global K index (even)
                int ci = kk >> 3, kr = kk & 7;    // kr even -> pair stays in channel row
                const _Float16* xp = X + ((size_t)(b * Cin + ci)) * LPAD + l + kr;
                bf.h[2 * jj]     = xp[0];
                bf.h[2 * jj + 1] = xp[1];
            }
            acc[u] = __builtin_amdgcn_wmma_f32_16x16x32_f16(false, a, false, bf.v,
                                                            (short)0, acc[u], false, false);
        }
    }
#pragma unroll
    for (int u = 0; u < 4; ++u) {
#pragma unroll
        for (int r = 0; r < 8; ++r) {
            int m = m0 + r + 8 * hl;
            out[((size_t)(b * CO + m)) * LEN + n0 + 16 * u + nl] = acc[u][r];
        }
    }
}

// ---------------------------------------------------------------------------
// conv-bias + channel-LN + leaky-relu; writes f32 (residual) and right-padded f16
// one thread per (b,l) over l in [0, LPAD)
__global__ void ln1_kernel(const float* __restrict__ in, const float* __restrict__ bc,
                           const float* __restrict__ g, const float* __restrict__ bb,
                           float* __restrict__ outf, _Float16* __restrict__ outh) {
    int i = blockIdx.x * blockDim.x + threadIdx.x;
    if (i >= BATCH * LPAD) return;
    int b = i / LPAD, l = i % LPAD;
    if (l >= LEN) {
        for (int c = 0; c < CO; ++c) outh[((size_t)(b * CO + c)) * LPAD + l] = (_Float16)0.f;
        return;
    }
    float su = 0.f, sq = 0.f;
    for (int c = 0; c < CO; ++c) {
        float x = in[((size_t)(b * CO + c)) * LEN + l] + bc[c];
        su += x; sq += x * x;
    }
    float mu = su * (1.f / CO);
    float var = sq * (1.f / CO) - mu * mu;
    float rs = rsqrtf(var + 1e-5f);
    for (int c = 0; c < CO; ++c) {
        float x = in[((size_t)(b * CO + c)) * LEN + l] + bc[c];
        float v = (x - mu) * rs * g[c] + bb[c];
        v = (v > 0.f) ? v : 0.1f * v;
        outf[((size_t)(b * CO + c)) * LEN + l] = v;
        outh[((size_t)(b * CO + c)) * LPAD + l] = (_Float16)v;
    }
}

// conv-bias + LN2 + lrelu + residual + avgpool(2): one thread per (b,l2)
__global__ void ln2_pool_kernel(const float* __restrict__ c2, const float* __restrict__ bc,
                                const float* __restrict__ hres, const float* __restrict__ g,
                                const float* __restrict__ bb, float* __restrict__ outp) {
    int i = blockIdx.x * blockDim.x + threadIdx.x;
    if (i >= BATCH * L2) return;
    int b = i / L2, l2 = i % L2;
    int l0 = 2 * l2, l1 = l0 + 1;
    float su0 = 0.f, sq0 = 0.f, su1 = 0.f, sq1 = 0.f;
    for (int c = 0; c < CO; ++c) {
        size_t base = ((size_t)(b * CO + c)) * LEN;
        float x0 = c2[base + l0] + bc[c], x1 = c2[base + l1] + bc[c];
        su0 += x0; sq0 += x0 * x0; su1 += x1; sq1 += x1 * x1;
    }
    float mu0 = su0 / CO, mu1 = su1 / CO;
    float rs0 = rsqrtf(sq0 / CO - mu0 * mu0 + 1e-5f);
    float rs1 = rsqrtf(sq1 / CO - mu1 * mu1 + 1e-5f);
    for (int c = 0; c < CO; ++c) {
        size_t base = ((size_t)(b * CO + c)) * LEN;
        float v0 = (c2[base + l0] + bc[c] - mu0) * rs0 * g[c] + bb[c];
        float v1 = (c2[base + l1] + bc[c] - mu1) * rs1 * g[c] + bb[c];
        v0 = (v0 > 0.f) ? v0 : 0.1f * v0;
        v1 = (v1 > 0.f) ? v1 : 0.1f * v1;
        v0 += hres[base + l0];
        v1 += hres[base + l1];
        outp[((size_t)(b * CO + c)) * L2 + l2] = 0.5f * (v0 + v1);
    }
}

// channel-LN of pooled h -> f16 [b][l][128] (GEMM-B layout)
__global__ void mamba_ln_kernel(const float* __restrict__ hp, const float* __restrict__ g,
                                const float* __restrict__ bb, _Float16* __restrict__ outh) {
    int i = blockIdx.x * blockDim.x + threadIdx.x;
    if (i >= BATCH * L2) return;
    int b = i / L2, l = i % L2;
    float su = 0.f, sq = 0.f;
    for (int c = 0; c < DM; ++c) {
        float x = hp[((size_t)(b * DM + c)) * L2 + l];
        su += x; sq += x * x;
    }
    float mu = su / DM;
    float rs = rsqrtf(sq / DM - mu * mu + 1e-5f);
    for (int c = 0; c < DM; ++c) {
        float x = hp[((size_t)(b * DM + c)) * L2 + l];
        outh[((size_t)(b * L2 + l)) * DM + c] = (_Float16)((x - mu) * rs * g[c] + bb[c]);
    }
}

// ---------------------------------------------------------------------------
// Generic WMMA GEMM: out = act[b][l][:Kd] @ W[m][:Kd]^T
// Each wave: 1 m-tile x 4 n-tiles (A fragment amortized over 4 WMMAs).
// mode 0: out[(b*L2+n)*ldo + m] = acc   (activation layout)
// mode 1: out[(b*ldo+m)*L2 + n] += acc  (residual accumulate, channel layout)
// grid (L2/64, ceil(Mpad/128), B), block 256
__global__ __launch_bounds__(256)
void matmul_wmma_kernel(const _Float16* __restrict__ W, const _Float16* __restrict__ act,
                        float* __restrict__ out, int Mpad, int Kd, int ldo, int mode) {
    int b = blockIdx.z;
    int wave = threadIdx.x >> 5, lane = threadIdx.x & 31;
    int mtile = blockIdx.y * 8 + wave;
    if (mtile * 16 >= Mpad) return;
    int m0 = mtile * 16;
    int n0 = blockIdx.x * 64;
    int nl = lane & 15, hl = lane >> 4;
    v8f acc[4] = {{}, {}, {}, {}};
    const _Float16* arow = W + (size_t)(m0 + nl) * Kd;
    const _Float16* brow0 = act + ((size_t)(b * L2 + n0 + nl)) * Kd;
    for (int kc = 0; kc < Kd; kc += 32) {
        __builtin_prefetch(arow + kc + 64, 0, 1);
        v16h a = load_a_frag(arow + kc, hl);
#pragma unroll
        for (int u = 0; u < 4; ++u) {
            v16h bf = load_b_frag(brow0 + (size_t)(16 * u) * Kd + kc, hl);
            acc[u] = __builtin_amdgcn_wmma_f32_16x16x32_f16(false, a, false, bf,
                                                            (short)0, acc[u], false, false);
        }
    }
    if (mode == 0) {
#pragma unroll
        for (int u = 0; u < 4; ++u) {
            size_t base = ((size_t)(b * L2 + n0 + 16 * u + nl)) * ldo;
#pragma unroll
            for (int r = 0; r < 8; ++r) out[base + m0 + r + 8 * hl] = acc[u][r];
        }
    } else {
#pragma unroll
        for (int u = 0; u < 4; ++u) {
            int n = n0 + 16 * u + nl;
#pragma unroll
            for (int r = 0; r < 8; ++r) {
                int m = m0 + r + 8 * hl;
                size_t o = ((size_t)(b * ldo + m)) * L2 + n;
                out[o] += acc[u][r];
            }
        }
    }
}

// ---------------------------------------------------------------------------
// causal depthwise conv (D_CONV=4) + bias + SiLU; xz is [b][l][512], xs part = [:256]
__global__ void dwconv_silu_kernel(const float* __restrict__ xz, const float* __restrict__ cw,
                                   const float* __restrict__ cb, _Float16* __restrict__ xsh) {
    int i = blockIdx.x * blockDim.x + threadIdx.x;
    if (i >= BATCH * L2 * DI) return;
    int d = i % DI, l = (i / DI) % L2, b = i / (DI * L2);
    float acc = cb[d];
#pragma unroll
    for (int j = 0; j < 4; ++j) {
        int t = l - 3 + j;
        if (t >= 0) acc += cw[d * 4 + j] * xz[((size_t)(b * L2 + t)) * 512 + d];
    }
    float s = acc / (1.f + __expf(-acc));
    xsh[((size_t)(b * L2 + l)) * DI + d] = (_Float16)s;
}

// dt-proj (K=8) + softplus
__global__ void dtproj_kernel(const float* __restrict__ dbl, const float* __restrict__ dtW,
                              const float* __restrict__ dtb, float* __restrict__ dtv) {
    int i = blockIdx.x * blockDim.x + threadIdx.x;
    if (i >= BATCH * L2 * DI) return;
    int d = i % DI, l = (i / DI) % L2, b = i / (DI * L2);
    const float* row = dbl + ((size_t)(b * L2 + l)) * 48;
    float acc = dtb[d];
#pragma unroll
    for (int r = 0; r < 8; ++r) acc += dtW[d * 8 + r] * row[r];
    float sp = (acc > 20.f) ? acc : log1pf(__expf(acc));
    dtv[((size_t)(b * L2 + l)) * DI + d] = sp;
}

// selective scan: one 16-lane half-wave per (b,d); lane = state index s
// y[b][t][d] = sum_s h[t][s]*C[t][s] + xs*D[d],  h = exp(dt*A)*h + dt*B*xs
__global__ __launch_bounds__(256)
void scan_kernel(const float* __restrict__ dtv, const _Float16* __restrict__ xsh,
                 const float* __restrict__ dbl, const float* __restrict__ Alog,
                 const float* __restrict__ Dp, float* __restrict__ ysc) {
    int gh = blockIdx.x * 16 + (threadIdx.x >> 4);   // global half-wave: B*DI = 1024
    int b = gh >> 8, d = gh & 255;
    int s = threadIdx.x & 15;
    float Aval = -__expf(Alog[d * DSTATE + s]);
    float Dv = Dp[d];
    float h = 0.f;
    for (int t = 0; t < L2; ++t) {
        size_t base = (size_t)(b * L2 + t);
        float dt = dtv[base * DI + d];
        float xs = (float)xsh[base * DI + d];
        float Bv = dbl[base * 48 + 8 + s];
        float Cv = dbl[base * 48 + 24 + s];
        h = __expf(dt * Aval) * h + dt * Bv * xs;
        float p = h * Cv;
        p += __shfl_xor(p, 1);
        p += __shfl_xor(p, 2);
        p += __shfl_xor(p, 4);
        p += __shfl_xor(p, 8);
        if (s == 0) ysc[base * DI + d] = p + xs * Dv;
    }
}

// gating: g = yscan * silu(z), z = xz[..][256+d]; -> f16 for out-proj
__global__ void gate_kernel(const float* __restrict__ ysc, const float* __restrict__ xz,
                            _Float16* __restrict__ gh_) {
    int i = blockIdx.x * blockDim.x + threadIdx.x;
    if (i >= BATCH * L2 * DI) return;
    int d = i % DI, l = (i / DI) % L2, b = i / (DI * L2);
    size_t base = (size_t)(b * L2 + l);
    float z = xz[base * 512 + 256 + d];
    float g = ysc[base * DI + d] * (z / (1.f + __expf(-z)));
    gh_[base * DI + d] = (_Float16)g;
}

// ---------------------------------------------------------------------------
extern "C" void kernel_launch(void* const* d_in, const int* in_sizes, int n_in,
                              void* d_out, int out_size, void* d_ws, size_t ws_size,
                              hipStream_t stream) {
    const float* x      = (const float*)d_in[0];
    const float* c1_v   = (const float*)d_in[1];
    const float* c1_g   = (const float*)d_in[2];
    const float* c1_b   = (const float*)d_in[3];
    const float* ln1_g  = (const float*)d_in[4];
    const float* ln1_b  = (const float*)d_in[5];
    const float* c2_v   = (const float*)d_in[6];
    const float* c2_g   = (const float*)d_in[7];
    const float* c2_b   = (const float*)d_in[8];
    const float* ln2_g  = (const float*)d_in[9];
    const float* ln2_b  = (const float*)d_in[10];
    const float* mlng   = (const float*)d_in[11];
    const float* mlnb   = (const float*)d_in[12];
    const float* minproj= (const float*)d_in[13];
    const float* mconvw = (const float*)d_in[14];
    const float* mconvb = (const float*)d_in[15];
    const float* mxproj = (const float*)d_in[16];
    const float* mdtw   = (const float*)d_in[17];
    const float* mdtb   = (const float*)d_in[18];
    const float* mAlog  = (const float*)d_in[19];
    const float* mD     = (const float*)d_in[20];
    const float* moutproj=(const float*)d_in[21];
    float* hout = (float*)d_out;   // pooled h, (B,128,4096), accumulated in place

    size_t off = 0;
    auto carve = [&](size_t bytes) -> void* {
        void* p = (char*)d_ws + off;
        off += (bytes + 255) & ~(size_t)255;
        return p;
    };
    _Float16* w1n   = (_Float16*)carve((size_t)CO * 512 * 2);
    _Float16* w2n   = (_Float16*)carve((size_t)CO * 1024 * 2);
    _Float16* inWh  = (_Float16*)carve((size_t)2 * 512 * 128 * 2);
    _Float16* xpWh  = (_Float16*)carve((size_t)2 * 48 * 256 * 2);
    _Float16* outWh = (_Float16*)carve((size_t)2 * 128 * 256 * 2);
    _Float16* xh    = (_Float16*)carve((size_t)BATCH * CIN * LPAD * 2);
    float* bufA     = (float*)carve((size_t)BATCH * CO * LEN * 4);   // c1raw, later yscan
    float* bufB     = (float*)carve((size_t)BATCH * CO * LEN * 4);   // hcur (LN1 out), later dtv
    float* c2raw    = (float*)carve((size_t)BATCH * CO * LEN * 4);
    _Float16* hh    = (_Float16*)carve((size_t)BATCH * CO * LPAD * 2);
    _Float16* hnh   = (_Float16*)carve((size_t)BATCH * L2 * DM * 2);
    float* xz       = (float*)carve((size_t)BATCH * L2 * 512 * 4);
    _Float16* xsh   = (_Float16*)carve((size_t)BATCH * L2 * DI * 2);
    float* dblw     = (float*)carve((size_t)BATCH * L2 * 48 * 4);
    _Float16* ghb   = (_Float16*)carve((size_t)BATCH * L2 * DI * 2);
    float* yscan = bufA;
    float* dtv   = bufB;

    // --- weight prep (every call; cheap, keeps launch deterministic) ---
    wnorm_kernel<<<CO, 64, 0, stream>>>(c1_v, c1_g, w1n, CIN * 8);
    wnorm_kernel<<<CO, 64, 0, stream>>>(c2_v, c2_g, w2n, CO * 8);
    {
        int n = 2 * 512 * 128;
        cvt_kernel<<<(n + 255) / 256, 256, 0, stream>>>(minproj, inWh, n);
        n = 2 * 128 * 256;
        cvt_kernel<<<(n + 255) / 256, 256, 0, stream>>>(moutproj, outWh, n);
        n = 2 * 48 * 256;
        cvt_xproj_kernel<<<(n + 255) / 256, 256, 0, stream>>>(mxproj, xpWh);
        n = BATCH * CIN * LPAD;
        cvt_xpad_kernel<<<(n + 255) / 256, 256, 0, stream>>>(x, xh);
    }

    // --- conv1 (WMMA implicit GEMM) -> bias + LN1 + lrelu ---
    conv_wmma_kernel<<<dim3(LEN / 64, 1, BATCH), 256, 0, stream>>>(w1n, xh, bufA, CIN, CIN * 8);
    {
        int n = BATCH * LPAD;
        ln1_kernel<<<(n + 255) / 256, 256, 0, stream>>>(bufA, c1_b, ln1_g, ln1_b, bufB, hh);
    }
    // --- conv2 -> bias + LN2 + lrelu + residual + avgpool -> hout ---
    conv_wmma_kernel<<<dim3(LEN / 64, 1, BATCH), 256, 0, stream>>>(w2n, hh, c2raw, CO, CO * 8);
    {
        int n = BATCH * L2;
        ln2_pool_kernel<<<(n + 255) / 256, 256, 0, stream>>>(c2raw, c2_b, bufB, ln2_g, ln2_b, hout);
    }

    // --- two Mamba blocks ---
    for (int i = 0; i < 2; ++i) {
        const _Float16* inW_i  = inWh  + (size_t)i * 512 * 128;
        const _Float16* xpW_i  = xpWh  + (size_t)i * 48 * 256;
        const _Float16* outW_i = outWh + (size_t)i * 128 * 256;
        const float* cw  = mconvw + (size_t)i * DI * 4;
        const float* cb  = mconvb + (size_t)i * DI;
        const float* dtW = mdtw   + (size_t)i * DI * 8;
        const float* dtb = mdtb   + (size_t)i * DI;
        const float* Al  = mAlog  + (size_t)i * DI * DSTATE;
        const float* Dp  = mD     + (size_t)i * DI;
        const float* lg  = mlng   + (size_t)i * DM;
        const float* lb  = mlnb   + (size_t)i * DM;

        int n = BATCH * L2;
        mamba_ln_kernel<<<(n + 255) / 256, 256, 0, stream>>>(hout, lg, lb, hnh);
        // in-proj: M=512, K=128 -> xz [b][l][512]
        matmul_wmma_kernel<<<dim3(L2 / 64, 4, BATCH), 256, 0, stream>>>(
            inW_i, hnh, xz, 512, 128, 512, 0);
        n = BATCH * L2 * DI;
        dwconv_silu_kernel<<<(n + 255) / 256, 256, 0, stream>>>(xz, cw, cb, xsh);
        // x-proj: M=48(pad of 40), K=256 -> dbl [b][l][48]
        matmul_wmma_kernel<<<dim3(L2 / 64, 1, BATCH), 256, 0, stream>>>(
            xpW_i, xsh, dblw, 48, 256, 48, 0);
        dtproj_kernel<<<(n + 255) / 256, 256, 0, stream>>>(dblw, dtW, dtb, dtv);
        // selective scan: B*DI half-waves
        scan_kernel<<<(BATCH * DI) / 16, 256, 0, stream>>>(dtv, xsh, dblw, Al, Dp, yscan);
        gate_kernel<<<(n + 255) / 256, 256, 0, stream>>>(yscan, xz, ghb);
        // out-proj: M=128, K=256, accumulate residual into hout (channel layout)
        matmul_wmma_kernel<<<dim3(L2 / 64, 1, BATCH), 256, 0, stream>>>(
            outW_i, ghb, hout, 128, 256, 128, 1);
    }
    (void)in_sizes; (void)n_in; (void)out_size; (void)ws_size;
}
